// LocalWindowAttention_40724879900952
// MI455X (gfx1250) — matrix operand order repaired and checked by reading, not compiled, not run on hardware
//
#include <hip/hip_runtime.h>
#include <hip/hip_bf16.h>
#include <stdint.h>

#define D_MODEL 1024
#define NHEAD   16
#define DH      64
#define WIN     128
#define SEQ     4096
#define BATCH   4
#define NB      (SEQ / WIN)     // 32
#define QKV_ROW (3 * D_MODEL)   // 3072

typedef __attribute__((ext_vector_type(16))) __bf16 v16bf;
typedef __attribute__((ext_vector_type(8)))  float  v8f;
typedef __attribute__((ext_vector_type(4)))  unsigned int u32x4;
typedef __attribute__((ext_vector_type(8)))  unsigned int u32x8;

struct Bytes32 { uint4 lo; uint4 hi; };

__device__ __forceinline__ unsigned short f2bf(float f) {
  unsigned int u = __builtin_bit_cast(unsigned int, f);
  u += 0x7FFFu + ((u >> 16) & 1u);          // round-to-nearest-even
  return (unsigned short)(u >> 16);
}

__device__ __forceinline__ v16bf load_frag2(const unsigned short* plo,
                                            const unsigned short* phi) {
  Bytes32 b;
  b.lo = *reinterpret_cast<const uint4*>(plo);
  b.hi = *reinterpret_cast<const uint4*>(phi);
  return __builtin_bit_cast(v16bf, b);
}

__device__ __forceinline__ v8f wmma_bf16(v16bf a, v16bf b, v8f c) {
  return __builtin_amdgcn_wmma_f32_16x16x32_bf16(false, a, false, b,
                                                 (short)0, c, false, false);
}

__device__ __forceinline__ v8f zero_v8f() {
  v8f z;
  #pragma unroll
  for (int i = 0; i < 8; ++i) z[i] = 0.0f;
  return z;
}

__device__ __forceinline__ void store_out(float* p, float v) { *p = v; }
__device__ __forceinline__ void store_out(unsigned short* p, float v) { *p = f2bf(v); }

__device__ __forceinline__ unsigned lds_off32(const void* p) {
  return (unsigned)(uintptr_t)p;   // LDS addr = generic addr[31:0]
}

// Async 16B global -> LDS copy (per-lane), tracked by ASYNCcnt.
__device__ __forceinline__ void async_b128(unsigned lds_off, const unsigned short* g) {
  unsigned long long ga = (unsigned long long)(uintptr_t)g;
  asm volatile("global_load_async_to_lds_b128 %0, %1, off"
               :: "v"(lds_off), "v"(ga) : "memory");
}

// Tensor Data Mover: 2D tile (tile_d0 elems x tile_d1 rows, 2-byte elements)
// from global (row stride d0_stride elems) into LDS, with LDS row padding.
// Descriptor per CDNA5 ISA 8.3/8.4 (groups 0 and 1; VADDR2/3 null -> 2D).
// Issued once per wave (TDM ignores EXEC); tracked by TENSORcnt.
__device__ __forceinline__ void tdm_load_2d(
    unsigned lds_addr, const void* gaddr,
    unsigned tile_d0, unsigned tile_d1, unsigned d0_stride,
    unsigned pad_interval_code, unsigned pad_amount_code) {
  unsigned long long ga = (unsigned long long)(uintptr_t)gaddr;
  u32x4 g0;
  g0[0] = 1u;                                              // count=1, user mode
  g0[1] = lds_addr;                                        // LDS byte address
  g0[2] = (unsigned)(ga & 0xFFFFFFFFull);                  // global_addr[31:0]
  g0[3] = (unsigned)((ga >> 32) & 0x1FFFFFFull) | (2u << 30);  // [56:32] | type=2
  u32x8 g1;
  g1[0] = (1u << 16)                                       // data_size = 2 bytes
        | (1u << 20)                                       // pad_enable
        | (pad_interval_code << 22) | (pad_amount_code << 25);
  g1[1] = (tile_d0 & 0xFFFFu) << 16;                       // tensor_dim0[15:0]
  g1[2] = (tile_d0 >> 16) | ((tile_d1 & 0xFFFFu) << 16);   // dim0 hi | tensor_dim1 lo
  g1[3] = ((tile_d1 >> 16) & 0xFFFFu) | (tile_d0 << 16);   // dim1 hi | tile_dim0
  g1[4] = tile_d1 & 0xFFFFu;                               // tile_dim1 (tile_dim2=0)
  g1[5] = d0_stride;                                       // tensor_dim0_stride lo
  g1[6] = 0u;                                              // stride hi | dim1_stride lo
  g1[7] = 0u;
  asm volatile("tensor_load_to_lds %0, %1" :: "s"(g0), "s"(g1) : "memory");
}

// ---------------------------------------------------------------------------
// bf16 GEMM with async-LDS double buffering.
//   C(MxN) = A(MxK,bf16 row-major) @ Bt(NxK,bf16 row-major = B transposed) + bias
// 128x128 tile / 256-thread WG (8 waves). K staged 32 at a time, 2 LDS buffers.
// LDS row stride 40 halves (80B): bank step 20 -> conflict-free frag reads.
// ---------------------------------------------------------------------------
template <typename TOUT>
__global__ __launch_bounds__(256) void gemm_bf16_async_wmma(
    const unsigned short* __restrict__ A, const unsigned short* __restrict__ Bt,
    const float* __restrict__ bias, TOUT* __restrict__ C,
    int M, int N, int K) {
  __shared__ unsigned short As[2][128 * 40];
  __shared__ unsigned short Bts[2][128 * 40];

  const int tid   = threadIdx.x;
  const int lane  = tid & 31;
  const int l16   = lane & 15;
  const int khalf = lane >> 4;
  const int w     = tid >> 5;
  const int wm    = (w & 1) * 64;
  const int wn    = (w >> 1) * 32;
  const int bm    = blockIdx.x * 128;
  const int bn    = blockIdx.y * 128;

  v8f acc[4][2];
  #pragma unroll
  for (int i = 0; i < 4; ++i)
    #pragma unroll
    for (int j = 0; j < 2; ++j) acc[i][j] = zero_v8f();

  const int nstages = K >> 5;

  // issue one stage (4 async b128 per thread: 2 for A, 2 for B)
  auto issue = [&](int ks, int buf) {
    const int k0 = ks << 5;
    #pragma unroll
    for (int i = 0; i < 2; ++i) {
      int e   = tid + (i << 8);        // 512 chunks: 128 rows x 4
      int row = e >> 2;
      int c   = e & 3;
      async_b128(lds_off32(&As[buf][row * 40 + c * 8]),
                 A + (size_t)(bm + row) * K + k0 + c * 8);
    }
    #pragma unroll
    for (int i = 0; i < 2; ++i) {
      int e   = tid + (i << 8);
      int col = e >> 2;
      int c   = e & 3;
      async_b128(lds_off32(&Bts[buf][col * 40 + c * 8]),
                 Bt + (size_t)(bn + col) * K + k0 + c * 8);
    }
  };

  issue(0, 0);
  for (int ks = 0; ks < nstages; ++ks) {
    const int buf = ks & 1;
    if (ks + 1 < nstages) {
      issue(ks + 1, buf ^ 1);
      asm volatile("s_wait_asynccnt 0x4" ::: "memory");  // stage ks complete
    } else {
      asm volatile("s_wait_asynccnt 0x0" ::: "memory");
    }
    __syncthreads();

    v16bf af[4];
    #pragma unroll
    for (int mt = 0; mt < 4; ++mt) {
      const unsigned short* p = &As[buf][(wm + mt * 16 + l16) * 40 + khalf * 8];
      af[mt] = load_frag2(p, p + 16);        // A layout: chunks khalf*8, +16
    }
    v16bf bfr[2];
    #pragma unroll
    for (int nt = 0; nt < 2; ++nt) {
      const unsigned short* p = &Bts[buf][(wn + nt * 16 + l16) * 40 + khalf * 8];
      bfr[nt] = load_frag2(p, p + 16);       // B layout: chunks khalf*8, +16
    }
    #pragma unroll
    for (int mt = 0; mt < 4; ++mt)
      #pragma unroll
      for (int nt = 0; nt < 2; ++nt)
        acc[mt][nt] = wmma_bf16(af[mt], bfr[nt], acc[mt][nt]);

    __syncthreads();   // all waves done with buf before it is overwritten
  }

  // Epilogue: C layout M = r + 8*khalf, N = l16
  #pragma unroll
  for (int mt = 0; mt < 4; ++mt) {
    #pragma unroll
    for (int nt = 0; nt < 2; ++nt) {
      const int gcol = bn + wn + nt * 16 + l16;
      const float bv = bias[gcol];
      #pragma unroll
      for (int r = 0; r < 8; ++r) {
        const int grow = bm + wm + mt * 16 + r + khalf * 8;
        store_out(&C[(size_t)grow * N + gcol], acc[mt][nt][r] + bv);
      }
    }
  }
}

// ---------------------------------------------------------------------------
// Prep kernels: fp32 -> bf16 convert, and transpose-convert for weights.
// ---------------------------------------------------------------------------
__global__ __launch_bounds__(256) void convert_bf16_kernel(
    const float* __restrict__ in, unsigned short* __restrict__ out) {
  const int i4 = (blockIdx.x * 256 + threadIdx.x) << 2;
  const float4 v = *reinterpret_cast<const float4*>(in + i4);
  unsigned int p0 = (unsigned int)f2bf(v.x) | ((unsigned int)f2bf(v.y) << 16);
  unsigned int p1 = (unsigned int)f2bf(v.z) | ((unsigned int)f2bf(v.w) << 16);
  *reinterpret_cast<uint2*>(out + i4) = make_uint2(p0, p1);
}

// in: [K][N] fp32, out: [N][K] bf16  (K = 1024, power of two)
__global__ __launch_bounds__(256) void transpose_bf16_kernel(
    const float* __restrict__ in, unsigned short* __restrict__ out, int N) {
  const int idx = blockIdx.x * 256 + threadIdx.x;
  const int k = idx & (D_MODEL - 1);
  const int n = idx >> 10;
  out[idx] = f2bf(in[(size_t)k * N + n]);
}

// Transpose V out of qkv (bf16) into vt[b][h][d][s]
__global__ __launch_bounds__(256) void build_vt_kernel(
    const unsigned short* __restrict__ qkv, unsigned short* __restrict__ vt) {
  const int idx = blockIdx.x * 256 + threadIdx.x;
  const int s = idx & (SEQ - 1);
  const int d = (idx >> 12) & (DH - 1);
  const int h = (idx >> 18) & (NHEAD - 1);
  const int b = idx >> 22;
  vt[idx] = qkv[((size_t)(b * SEQ + s)) * QKV_ROW + 2 * D_MODEL + h * DH + d];
}

// ---------------------------------------------------------------------------
// Banded attention: one WG per (block n, head h, batch b). 8 waves x 16 rows.
// K window staged by the Tensor Data Mover: each wave TDM-loads a 48-row slice
// (clamped at sequence edges; garbage rows are finite bf16 and masked later).
// TDM pads LDS rows to 72 halves (bank step 36 -> conflict-free frag reads).
// ---------------------------------------------------------------------------
__global__ __launch_bounds__(256) void attn_kernel(
    const unsigned short* __restrict__ qkv,   // [B*S][3072] bf16 (q|k|v)
    const unsigned short* __restrict__ vt,    // [B][H][64][S] bf16
    unsigned short* __restrict__ attn_out) {  // [B*S][1024] bf16
  __shared__ unsigned short Ks[384 * 72];     // padded rows (written by TDM)
  __shared__ unsigned short Ps[8 * 16 * 40];  // per-wave P re-layout scratch

  const int n = blockIdx.x, h = blockIdx.y, b = blockIdx.z;
  const int tid   = threadIdx.x;
  const int lane  = tid & 31;
  const int l16   = lane & 15;
  const int khalf = lane >> 4;
  const int w     = tid >> 5;

  const size_t base_bs = (size_t)b * SEQ;
  const int kws = n * WIN - WIN;              // window start (can be negative)

  // TDM: wave w copies window rows [w*48, w*48+48) -> Ks with padded stride.
  // pad_interval code 4 = 32 DWORDs (64 halves), pad_amount code 3 = 4 DWORDs.
  {
    const int slice  = kws + w * 48;
    const int sstart = min(max(slice, 0), SEQ - 48);   // clamp: rows masked later
    tdm_load_2d(lds_off32(&Ks[w * 48 * 72]),
                qkv + (base_bs + sstart) * QKV_ROW + D_MODEL + h * DH,
                /*tile_d0=*/64, /*tile_d1=*/48, /*d0_stride=*/QKV_ROW,
                /*pad_interval=*/4u, /*pad_amount=*/3u);
  }

  // Q A-fragments directly from global (overlap with TDM copy)
  const unsigned short* qrow =
      qkv + (base_bs + n * WIN + w * 16 + l16) * QKV_ROW + h * DH;
  v16bf qf[2];
  #pragma unroll
  for (int kc = 0; kc < 2; ++kc) {
    const unsigned short* p = qrow + kc * 32 + khalf * 8;
    qf[kc] = load_frag2(p, p + 16);
  }

  // Prefetch the Vt rows this wave will read in the PV phase.
  const unsigned short* vtb = vt + ((size_t)(b * NHEAD + h) * DH) * SEQ;
  const int pfs = min(max(kws, 0), SEQ - 16);
  #pragma unroll
  for (int nt = 0; nt < 4; ++nt)
    __builtin_prefetch(vtb + (size_t)(nt * 16 + l16) * SEQ + pfs, 0, 0);

  __builtin_amdgcn_s_wait_tensorcnt(0);
  __syncthreads();

  // Scores: 16x384 strip per wave = 24 f32 accumulator tiles
  v8f sc[24];
  #pragma unroll
  for (int kt = 0; kt < 24; ++kt) sc[kt] = zero_v8f();
  #pragma unroll
  for (int kt = 0; kt < 24; ++kt) {
    const int key = kt * 16 + l16;
    #pragma unroll
    for (int kc = 0; kc < 2; ++kc) {
      const unsigned short* plo = &Ks[key * 72 + kc * 32 + khalf * 8];
      sc[kt] = wmma_bf16(qf[kc], load_frag2(plo, plo + 16), sc[kt]);
    }
  }

  // Band mask + softmax (row layout: M = r + 8*khalf, N = l16)
  const float scale = 0.03125f;               // 1/sqrt(d_model)
  float rmax[8], rsum[8];
  #pragma unroll
  for (int r = 0; r < 8; ++r) rmax[r] = -3.0e38f;
  #pragma unroll
  for (int kt = 0; kt < 24; ++kt) {
    const int kj   = kt * 16 + l16;
    const int kpos = kws + kj;
    const bool kvalid = (kpos >= 0) && (kpos < SEQ);
    #pragma unroll
    for (int r = 0; r < 8; ++r) {
      const int qi = w * 16 + r + khalf * 8;
      const bool ok = kvalid && (kj >= qi) && (kj <= qi + 2 * WIN);
      const float v = ok ? sc[kt][r] * scale : -1.0e30f;
      sc[kt][r] = v;
      rmax[r] = fmaxf(rmax[r], v);
    }
  }
  #pragma unroll
  for (int r = 0; r < 8; ++r) {
    #pragma unroll
    for (int m = 1; m < 16; m <<= 1)
      rmax[r] = fmaxf(rmax[r], __shfl_xor(rmax[r], m, 32));
    rsum[r] = 0.0f;
  }
  #pragma unroll
  for (int kt = 0; kt < 24; ++kt) {
    #pragma unroll
    for (int r = 0; r < 8; ++r) {
      const float p = __expf(sc[kt][r] - rmax[r]);
      sc[kt][r] = p;
      rsum[r] += p;
    }
  }
  #pragma unroll
  for (int r = 0; r < 8; ++r) {
    #pragma unroll
    for (int m = 1; m < 16; m <<= 1)
      rsum[r] += __shfl_xor(rsum[r], m, 32);
  }

  // O = P @ V : bounce P (C-layout) through per-wave LDS into A-layout,
  // V B-fragments are contiguous loads from global vt.
  v8f o[4];
  #pragma unroll
  for (int nt = 0; nt < 4; ++nt) o[nt] = zero_v8f();
  unsigned short* ps = &Ps[w * 16 * 40];
  #pragma unroll
  for (int kk = 0; kk < 12; ++kk) {
    #pragma unroll
    for (int t = 0; t < 2; ++t)
      #pragma unroll
      for (int r = 0; r < 8; ++r)
        ps[(r + khalf * 8) * 40 + t * 16 + l16] = f2bf(sc[kk * 2 + t][r]);
    asm volatile("s_wait_dscnt 0" ::: "memory");   // P stores visible before reload
    const unsigned short* pa = ps + l16 * 40 + khalf * 16;
    v16bf af = load_frag2(pa, pa + 8);
    #pragma unroll
    for (int nt = 0; nt < 4; ++nt) {
      int s0 = kws + kk * 32 + khalf * 16;
      s0 = min(max(s0, 0), SEQ - 16);              // OOB rows have P==0; clamp = safe
      const unsigned short* pv = vtb + (size_t)(nt * 16 + l16) * SEQ + s0;
      o[nt] = wmma_bf16(af, load_frag2(pv, pv + 8), o[nt]);
    }
  }

  // Normalize and write bf16 attention output (B,S,1024)
  #pragma unroll
  for (int r = 0; r < 8; ++r) {
    const float inv = 1.0f / rsum[r];
    const int qpos  = n * WIN + w * 16 + r + khalf * 8;
    unsigned short* orow = attn_out + (base_bs + qpos) * D_MODEL + h * DH;
    #pragma unroll
    for (int nt = 0; nt < 4; ++nt)
      orow[nt * 16 + l16] = f2bf(o[nt][r] * inv);
  }
}

// ---------------------------------------------------------------------------
extern "C" void kernel_launch(void* const* d_in, const int* in_sizes, int n_in,
                              void* d_out, int out_size, void* d_ws, size_t ws_size,
                              hipStream_t stream) {
  const float* x    = (const float*)d_in[0];
  const float* Wqkv = (const float*)d_in[1];
  const float* bqkv = (const float*)d_in[2];
  const float* Wout = (const float*)d_in[3];
  const float* bout = (const float*)d_in[4];
  float* out = (float*)d_out;

  char* ws = (char*)d_ws;
  unsigned short* xb     = (unsigned short*)ws;                    // 32 MB
  unsigned short* wqkvT  = (unsigned short*)(ws + 33554432);       //  6 MB
  unsigned short* woutT  = (unsigned short*)(ws + 39845888);       //  2 MB
  unsigned short* qkvb   = (unsigned short*)(ws + 41943040);       // 96 MB
  unsigned short* vtb    = (unsigned short*)(ws + 142606336);      // 32 MB
  unsigned short* attnb  = (unsigned short*)(ws + 176160768);      // 32 MB

  const int Mrows = BATCH * SEQ;  // 16384

  // fp32 -> bf16 prep (x), transposed bf16 weights (Wqkv^T, Wout^T)
  convert_bf16_kernel<<<(Mrows * D_MODEL) / 1024, 256, 0, stream>>>(x, xb);
  transpose_bf16_kernel<<<(QKV_ROW * D_MODEL) / 256, 256, 0, stream>>>(Wqkv, wqkvT, QKV_ROW);
  transpose_bf16_kernel<<<(D_MODEL * D_MODEL) / 256, 256, 0, stream>>>(Wout, woutT, D_MODEL);

  dim3 g1(Mrows / 128, QKV_ROW / 128);
  gemm_bf16_async_wmma<unsigned short><<<g1, 256, 0, stream>>>(
      xb, wqkvT, bqkv, qkvb, Mrows, QKV_ROW, D_MODEL);

  build_vt_kernel<<<(BATCH * NHEAD * DH * SEQ) / 256, 256, 0, stream>>>(qkvb, vtb);

  dim3 ga(NB, NHEAD, BATCH);
  attn_kernel<<<ga, 256, 0, stream>>>(qkvb, vtb, attnb);

  dim3 g2(Mrows / 128, D_MODEL / 128);
  gemm_bf16_async_wmma<float><<<g2, 256, 0, stream>>>(
      attnb, woutT, bout, out, Mrows, D_MODEL, D_MODEL);
}
